// CFTransformerEncoderVN_83786222011229
// MI455X (gfx1250) — matrix operand order repaired and checked by reading, not compiled, not run on hardware
//
#include <hip/hip_runtime.h>
#include <hip/hip_bf16.h>
#include <stdint.h>

// ---------------------------------------------------------------------------
// CDNA5 (gfx1250) implementation: all dense linear algebra routed through
// v_wmma_f32_16x16x32_bf16 (bf16 operands, f32 accumulate), weight tiles
// fetched by the Tensor Data Mover (tensor_load_to_lds, 6-arg clang-23 form).
// wave32.
// ---------------------------------------------------------------------------

#define SCA 256
#define VEC 64
#define EC  64
#define NLAYER 6
#define PI_F 3.14159265358979f

typedef __bf16 bf16_t;
typedef __attribute__((ext_vector_type(16))) __bf16 v16bf;
typedef __attribute__((ext_vector_type(8)))  __bf16 v8bf;
typedef __attribute__((ext_vector_type(8)))  float  v8f;
typedef __attribute__((ext_vector_type(4)))  unsigned int v4u;
typedef __attribute__((ext_vector_type(8)))  int v8i;
typedef __attribute__((ext_vector_type(4)))  int v4i;

#if defined(__gfx1250__) && __has_builtin(__builtin_amdgcn_tensor_load_to_lds) && \
    __has_builtin(__builtin_amdgcn_s_wait_tensorcnt)
#define USE_TDM 1
#else
#define USE_TDM 0
#endif

__device__ __forceinline__ bf16_t f2bf(float f) {
  union { float f; uint32_t u; } in; in.f = f;
  uint32_t r = in.u + 0x7fffu + ((in.u >> 16) & 1u);   // round-to-nearest-even
  union { uint16_t u; bf16_t b; } out; out.u = (uint16_t)(r >> 16);
  return out.b;
}

// ---------------------------------------------------------------------------
// Generic WMMA GEMM:  Out[r, m] = epi( sum_k A[ar, k] * W[m, k] )
//   ar = aIdx ? aIdx[r] : r          (row gather, e.g. nsca[col[e]])
//   epi: +bias[m]; +addM[m]*addR[r]; act(0 none,1 leaky .01,2 sigmoid);
//        * mul[(mulIdx?mulIdx[r/mulDiv]:r/mulDiv)*mulLd + m]
// block = 256 threads (8 waves), tile 64(R) x 64(M), K step 64 (all K %64==0)
// Each wave owns two 16x16 tiles in the same row strip -> shared A fragment.
// Weight tile (64 x 64 f32, row stride ldw) DMA'd to LDS by the TDM.
// ---------------------------------------------------------------------------
__global__ __launch_bounds__(256)
void wmma_gemm_kernel(const float* __restrict__ A, int lda, const int* __restrict__ aIdx,
                      const float* __restrict__ W, int ldw, int K,
                      const float* __restrict__ bias,
                      const float* __restrict__ addM, const float* __restrict__ addR,
                      int act,
                      const float* __restrict__ mul, const int* __restrict__ mulIdx,
                      int mulDiv, int mulLd,
                      float* __restrict__ Out, int ldo, int M)
{
  __shared__ __align__(16) bf16_t As[64 * 64];
  __shared__ __align__(16) bf16_t Ws[64 * 64];
#if USE_TDM
  __shared__ __align__(16) float  Wst[64 * 64];
#endif

  const int n0  = blockIdx.x * 64;     // output column base
  const int r0  = blockIdx.y * 64;     // output row base
  const int tid = threadIdx.x;
  const int wave = tid >> 5;
  const int lane = tid & 31;
  const int half = lane >> 4;
  const int lr   = lane & 15;

  // wave -> row strip mi, column tiles ni0/ni1 (share one A fragment)
  const int mi  = wave >> 1;
  const int ni0 = (wave & 1) * 2;
  const int ni1 = ni0 + 1;

  v8f acc0 = {}; v8f acc1 = {};

  for (int k0 = 0; k0 < K; k0 += 64) {
#if USE_TDM
    // ---- Tensor Data Mover: DMA W tile [n0..n0+63, k0..k0+63] (f32) to LDS
    if (wave == 0) {
      uint64_t ga  = (uint64_t)(uintptr_t)(W + (size_t)n0 * ldw + k0);
      unsigned lds = (unsigned)(uintptr_t)(&Wst[0]);
      v4u g0;
      g0.x = 1u;                                   // count=1, user mode
      g0.y = lds;                                  // lds_addr
      g0.z = (unsigned)(ga & 0xffffffffu);         // global_addr[31:0]
      g0.w = (unsigned)((ga >> 32) & 0x01ffffffu)  // global_addr[56:32]
             | (2u << 30);                         // type = 2 ("image")
      unsigned td0 = (unsigned)ldw;                // tensor_dim0 (elements)
      unsigned td1 = 0x10000u;                     // tensor_dim1 (large)
      uint64_t st0 = (uint64_t)ldw;                // tensor_dim0_stride
      v8i g1;
      g1[0] = (int)(2u << 16);                     // data_size = 4B
      g1[1] = (int)((td0 & 0xffffu) << 16);        // tensor_dim0 lo
      g1[2] = (int)(((td0 >> 16) & 0xffffu) | ((td1 & 0xffffu) << 16));
      g1[3] = (int)(((td1 >> 16) & 0xffffu) | (64u << 16));   // tile_dim0 = 64
      g1[4] = (int)(64u);                          // tile_dim1 = 64
      g1[5] = (int)(st0 & 0xffffffffu);
      g1[6] = (int)((st0 >> 32) & 0xffffu);
      g1[7] = 0;
      v4i z4 = {0, 0, 0, 0};
      v8i z8 = {0, 0, 0, 0, 0, 0, 0, 0};
      __builtin_amdgcn_tensor_load_to_lds(g0, g1, z4, z4, z8, 0);
    }
#endif
    // ---- stage A tile (64 rows x 64 k) as bf16 (with optional row gather)
    for (int i = tid; i < 64 * 64; i += 256) {
      int rr = i >> 6, kk = i & 63;
      int gr = r0 + rr;
      int ar = aIdx ? aIdx[gr] : gr;
      As[i] = f2bf(A[(size_t)ar * lda + k0 + kk]);
    }
    // prefetch next A k-stage (lowers to global_prefetch_b8)
    if (k0 + 64 < K) {
      int i = tid * 16;
      int rr = i >> 6;
      int gr = r0 + rr;
      int ar = aIdx ? aIdx[gr] : gr;
      __builtin_prefetch(&A[(size_t)ar * lda + k0 + 64 + (i & 63)], 0, 1);
    }
#if USE_TDM
    if (wave == 0) __builtin_amdgcn_s_wait_tensorcnt(0);
    __syncthreads();
    for (int i = tid; i < 64 * 64; i += 256) Ws[i] = f2bf(Wst[i]);
#else
    for (int i = tid; i < 64 * 64; i += 256) {
      int nn = i >> 6, kk = i & 63;
      Ws[i] = f2bf(W[(size_t)(n0 + nn) * ldw + k0 + kk]);
    }
#endif
    __syncthreads();

    // A fragment layout (ISA 7.12.2, 16-bit A 16x32):
    //   lane(half,lr): row m=lr, k = (j<8?0:16) + half*8 + (j&7)
    // B fragment (32x16): lane(half,lr): col n=lr, k = (j<8?0:16) + half*8 + (j&7)
    union Frag { v16bf v; v8bf h[2]; };
    #pragma unroll
    for (int h2 = 0; h2 < 2; ++h2) {
      const int ko = h2 * 32;
      Frag af, b0, b1;
      const bf16_t* ab = &As[(mi * 16 + lr) * 64 + ko];
      af.h[0] = *(const v8bf*)(ab + half * 8);
      af.h[1] = *(const v8bf*)(ab + 16 + half * 8);
      const bf16_t* wb0 = &Ws[(ni0 * 16 + lr) * 64 + ko];
      b0.h[0] = *(const v8bf*)(wb0 + half * 8);
      b0.h[1] = *(const v8bf*)(wb0 + 16 + half * 8);
      const bf16_t* wb1 = &Ws[(ni1 * 16 + lr) * 64 + ko];
      b1.h[0] = *(const v8bf*)(wb1 + half * 8);
      b1.h[1] = *(const v8bf*)(wb1 + 16 + half * 8);
      acc0 = __builtin_amdgcn_wmma_f32_16x16x32_bf16(false, af.v, false, b0.v,
                                                     (short)0, acc0, false, false);
      acc1 = __builtin_amdgcn_wmma_f32_16x16x32_bf16(false, af.v, false, b1.v,
                                                     (short)0, acc1, false, false);
    }
    __syncthreads();
  }

  // epilogue.  D layout: lane(half,lr): col n=lr, vgpr r -> row m = r + 8*half
  #pragma unroll
  for (int t = 0; t < 2; ++t) {
    v8f acc = t ? acc1 : acc0;
    int ni = t ? ni1 : ni0;
    int gm = n0 + ni * 16 + lr;
    #pragma unroll
    for (int rg = 0; rg < 8; ++rg) {
      int gr = r0 + mi * 16 + rg + 8 * half;
      float v = acc[rg];
      if (bias) v += bias[gm];
      if (addM) v += addM[gm] * addR[gr];
      if (act == 1)      v = v > 0.f ? v : 0.01f * v;
      else if (act == 2) v = 1.f / (1.f + __expf(-v));
      if (mul) {
        int mr = gr / mulDiv;
        if (mulIdx) mr = mulIdx[mr];
        v *= mul[(size_t)mr * mulLd + gm];
      }
      Out[(size_t)gr * ldo + gm] = v;
    }
  }
}

// ---------------------------------------------------------------------------
// small elementwise / reduction kernels
// ---------------------------------------------------------------------------
__global__ void k_copy(const float* __restrict__ s, float* __restrict__ d, int n) {
  int i = blockIdx.x * blockDim.x + threadIdx.x; if (i < n) d[i] = s[i];
}
__global__ void k_zero(float* __restrict__ d, int n) {
  int i = blockIdx.x * blockDim.x + threadIdx.x; if (i < n) d[i] = 0.f;
}
__global__ void k_add(float* __restrict__ x, const float* __restrict__ d, int n) {
  int i = blockIdx.x * blockDim.x + threadIdx.x; if (i < n) x[i] += d[i];
}
// (n,o,x) -> internal (n*3+x, o):  vecT[n*192 + x*64 + o] = nv[n*192 + o*3 + x]
__global__ void k_vec_in(const float* __restrict__ nv, float* __restrict__ vecT, int N) {
  int i = blockIdx.x * blockDim.x + threadIdx.x;
  if (i >= N * 192) return;
  int n = i / 192, j = i % 192, x = j / 64, o = j % 64;
  vecT[i] = nv[n * 192 + o * 3 + x];
}
__global__ void k_out_pack(const float* __restrict__ sca, const float* __restrict__ vecT,
                           float* __restrict__ out, int N) {
  int i = blockIdx.x * blockDim.x + threadIdx.x;
  int total = N * (SCA + 192);
  if (i >= total) return;
  if (i < N * SCA) { out[i] = sca[i]; return; }
  int j = i - N * SCA;
  int n = j / 192, r = j % 192, o = r / 3, x = r % 3;
  out[i] = vecT[n * 192 + x * 64 + o];
}
__global__ void k_edge_geom(const int* __restrict__ ei, const float* __restrict__ pos, int E,
                            float* __restrict__ unit, float* __restrict__ Cc,
                            float* __restrict__ un, float* __restrict__ s2,
                            float* __restrict__ esca_g) {
  int e = blockIdx.x * blockDim.x + threadIdx.x;
  if (e >= E) return;
  int r = ei[e], c = ei[E + e];
  float ex = pos[r * 3]     - pos[c * 3];
  float ey = pos[r * 3 + 1] - pos[c * 3 + 1];
  float ez = pos[r * 3 + 2] - pos[c * 3 + 2];
  float dist = sqrtf(ex * ex + ey * ey + ez * ez);
  float inv = 1.f / (dist + 1e-7f);
  unit[e * 3]     = ex * inv;
  unit[e * 3 + 1] = ey * inv;
  unit[e * 3 + 2] = ez * inv;
  float uu = dist * inv;
  un[e] = uu; s2[e] = uu * uu;
  Cc[e] = (dist <= 10.f) ? 0.5f * (__cosf(dist * (PI_F / 10.f)) + 1.f) : 0.f;
  const float step = 10.f / 63.f;
  const float coeff = -0.5f / (step * step);
  for (int k = 0; k < EC; ++k) {
    float dlt = dist - (float)k * step;
    esca_g[(size_t)e * EC + k] = __expf(coeff * dlt * dlt);
  }
}
// fold rank-1 edge-vector structure: w_eff = wv1@ee_w, w2_eff = wv2@w_eff,
// A[o] = sum_j ws[o, j] * |w_eff[j]|  (first 64 cols of egv_ws, ld 128)
__global__ void k_layer_prep(const float* __restrict__ wv1, const float* __restrict__ wv2,
                             const float* __restrict__ eew, const float* __restrict__ ws,
                             float* __restrict__ prep) {
  __shared__ float sw[64];
  int o = threadIdx.x;
  float we = 0.f;
  for (int i = 0; i < 64; ++i) we += wv1[o * 64 + i] * eew[i];
  sw[o] = we;
  __syncthreads();
  float w2 = 0.f, av = 0.f;
  for (int i = 0; i < 64; ++i) w2 += wv2[o * 64 + i] * sw[i];
  for (int j = 0; j < 64; ++j) av += ws[o * 128 + j] * fabsf(sw[j]);
  prep[o] = sw[o]; prep[64 + o] = w2; prep[128 + o] = av;
}
__global__ void k_gmul(const float* __restrict__ gate, const float* __restrict__ prep,
                       float* __restrict__ g, int n) {
  int i = blockIdx.x * blockDim.x + threadIdx.x;
  if (i < n) g[i] = gate[i] * prep[64 + (i & 63)];
}
// colinear VN-LeakyReLU coefficients + leaky on esca (in place)
__global__ void k_f_lrelu(float* __restrict__ esca, const float* __restrict__ g,
                          const float* __restrict__ dd, const float* __restrict__ s2,
                          float* __restrict__ f, int E) {
  int i = blockIdx.x * blockDim.x + threadIdx.x;
  if (i >= E * 64) return;
  int e = i >> 6;
  float gg = g[i], dv = dd[i], ss = s2[e];
  float dot = gg * dv * ss;
  float fo;
  if (dot >= 0.f) fo = gg;
  else {
    float dsq = dv * dv * ss;
    fo = 0.2f * gg + 0.8f * (gg - (dot / (dsq + 1e-6f)) * dv);
  }
  f[i] = fo;
  float ev = esca[i];
  esca[i] = ev > 0.f ? ev : 0.01f * ev;
}
// cat[r,0:64] = vn(vi), cat[r,64:320] = sca[r]
__global__ void k_vn_pack(const float* __restrict__ vi, const float* __restrict__ sca,
                          float* __restrict__ cat, int R) {
  int i = blockIdx.x * blockDim.x + threadIdx.x;
  if (i >= R * 320) return;
  int r = i / 320, c = i % 320;
  if (c < 64) {
    float a = vi[(size_t)r * 192 + c];
    float b = vi[(size_t)r * 192 + 64 + c];
    float d = vi[(size_t)r * 192 + 128 + c];
    cat[i] = sqrtf(a * a + b * b + d * d);
  } else {
    cat[i] = sca[(size_t)r * 256 + (c - 64)];
  }
}
// y_vec[(3e+x),o] = t2*nvec_j + t3*(t4*unit + evn_b)
__global__ void k_yvec(const float* __restrict__ t2, const float* __restrict__ t3,
                       const float* __restrict__ t4, const float* __restrict__ nvecT,
                       const float* __restrict__ unit, const float* __restrict__ evn_b,
                       const int* __restrict__ col, float* __restrict__ yv, int E) {
  int i = blockIdx.x * blockDim.x + threadIdx.x;
  if (i >= E * 64) return;
  int e = i >> 6, o = i & 63;
  int cj = col[e];
  float a2 = t2[i], a3 = t3[i], a4 = t4[i], bb = evn_b[o];
  #pragma unroll
  for (int x = 0; x < 3; ++x) {
    float nv = nvecT[(size_t)cj * 192 + x * 64 + o];
    float evl = a4 * unit[e * 3 + x] + bb;
    yv[(size_t)e * 192 + x * 64 + o] = a2 * nv + a3 * evl;
  }
}
__global__ void k_scatter_sca(const float* __restrict__ m, const float* __restrict__ Cc,
                              const int* __restrict__ row, float* __restrict__ a, int E) {
  int i = blockIdx.x * blockDim.x + threadIdx.x;
  if (i >= E * SCA) return;
  int e = i / SCA, c = i % SCA;
  atomicAdd(&a[(size_t)row[e] * SCA + c], m[i] * Cc[e]);
}
__global__ void k_scatter_vec(const float* __restrict__ m, const float* __restrict__ Cc,
                              const int* __restrict__ row, float* __restrict__ a, int E) {
  int i = blockIdx.x * blockDim.x + threadIdx.x;
  if (i >= E * 192) return;
  int e = i / 192, j = i % 192;
  atomicAdd(&a[(size_t)row[e] * 192 + j], m[i] * Cc[e]);
}
// LayerNorm(256) + leaky, one node per block (256 threads)
__global__ void k_combine_sca(const float* __restrict__ c, const float* __restrict__ a,
                              const float* __restrict__ g, const float* __restrict__ b,
                              float* __restrict__ o) {
  __shared__ float red[256];
  __shared__ float mu, var;
  int n = blockIdx.x, t = threadIdx.x;
  float v = c[(size_t)n * 256 + t] + a[(size_t)n * 256 + t];
  red[t] = v; __syncthreads();
  for (int s = 128; s > 0; s >>= 1) { if (t < s) red[t] += red[t + s]; __syncthreads(); }
  if (t == 0) mu = red[0] / 256.f; __syncthreads();
  float dv = v - mu;
  red[t] = dv * dv; __syncthreads();
  for (int s = 128; s > 0; s >>= 1) { if (t < s) red[t] += red[t + s]; __syncthreads(); }
  if (t == 0) var = red[0] / 256.f; __syncthreads();
  float r = dv * rsqrtf(var + 1e-5f) * g[t] + b[t];
  o[(size_t)n * 256 + t] = r > 0.f ? r : 0.01f * r;
}
// LayerNorm over (64,3)=192 per node; internal elem idx t = x*64+o, g/b[o*3+x]
__global__ void k_combine_vec(const float* __restrict__ c, const float* __restrict__ a,
                              const float* __restrict__ g, const float* __restrict__ b,
                              float* __restrict__ o) {
  __shared__ float red[256];
  __shared__ float mu, var;
  int n = blockIdx.x, t = threadIdx.x;   // 192 threads
  float v = c[(size_t)n * 192 + t] + a[(size_t)n * 192 + t];
  red[t] = v; if (t < 64) red[192 + t] = 0.f;
  __syncthreads();
  for (int s = 128; s > 0; s >>= 1) { if (t < s) red[t] += red[t + s]; __syncthreads(); }
  if (t == 0) mu = red[0] / 192.f; __syncthreads();
  float dv = v - mu;
  red[t] = dv * dv; if (t < 64) red[192 + t] = 0.f;
  __syncthreads();
  for (int s = 128; s > 0; s >>= 1) { if (t < s) red[t] += red[t + s]; __syncthreads(); }
  if (t == 0) var = red[0] / 192.f; __syncthreads();
  int x = t / 64, oo = t % 64;
  o[(size_t)n * 192 + t] = dv * rsqrtf(var + 1e-5f) * g[oo * 3 + x] + b[oo * 3 + x];
}
// full VN-LeakyReLU on node vectors (in place), d = act_dir @ vec precomputed
__global__ void k_vnlrelu(float* __restrict__ v, const float* __restrict__ d, int N) {
  int i = blockIdx.x * blockDim.x + threadIdx.x;
  if (i >= N * 64) return;
  size_t base = (size_t)(i >> 6) * 192 + (i & 63);
  float x0 = v[base], x1 = v[base + 64], x2 = v[base + 128];
  float d0 = d[base], d1 = d[base + 64], d2 = d[base + 128];
  float dot = x0 * d0 + x1 * d1 + x2 * d2;
  if (dot < 0.f) {
    float dsq = d0 * d0 + d1 * d1 + d2 * d2;
    float t = dot / (dsq + 1e-6f);
    v[base]       = 0.2f * x0 + 0.8f * (x0 - t * d0);
    v[base + 64]  = 0.2f * x1 + 0.8f * (x1 - t * d1);
    v[base + 128] = 0.2f * x2 + 0.8f * (x2 - t * d2);
  }
}

// ---------------------------------------------------------------------------
// host side
// ---------------------------------------------------------------------------
static inline void gemm(hipStream_t st, const float* A, int lda, const int* aIdx,
                        const float* W, int ldw, int K,
                        const float* bias, const float* addM, const float* addR,
                        int act, const float* mul, const int* mulIdx, int mulDiv, int mulLd,
                        float* Out, int ldo, int R, int M) {
  dim3 g(M / 64, R / 64), b(256);
  wmma_gemm_kernel<<<g, b, 0, st>>>(A, lda, aIdx, W, ldw, K, bias, addM, addR,
                                    act, mul, mulIdx, mulDiv, mulLd, Out, ldo, M);
}
static inline int cdiv(int a, int b) { return (a + b - 1) / b; }

// gvlinear: out_sca = [vn(wv1@vec) | sca] @ ws.T ; out_vec = sigmoid(out_sca@wg.T+bg) * (wv2@vi)
static void gvlinear(hipStream_t st, const float* sca_in, const float* vecT_in, int R,
                     const float* wv1, const float* wv2, const float* ws,
                     const float* wg, const float* bg,
                     float* vi, float* cat, float* gate,
                     float* out_sca, float* out_vecT) {
  gemm(st, vecT_in, 64, nullptr, wv1, 64, 64, nullptr, nullptr, nullptr, 0,
       nullptr, nullptr, 1, 0, vi, 64, R * 3, 64);
  k_vn_pack<<<cdiv(R * 320, 256), 256, 0, st>>>(vi, sca_in, cat, R);
  gemm(st, cat, 320, nullptr, ws, 320, 320, nullptr, nullptr, nullptr, 0,
       nullptr, nullptr, 1, 0, out_sca, 256, R, 256);
  gemm(st, out_sca, 256, nullptr, wg, 256, 256, bg, nullptr, nullptr, 2,
       nullptr, nullptr, 1, 0, gate, 64, R, 64);
  gemm(st, vi, 64, nullptr, wv2, 64, 64, nullptr, nullptr, nullptr, 0,
       gate, nullptr, 3, 64, out_vecT, 64, R * 3, 64);
}

extern "C" void kernel_launch(void* const* d_in, const int* in_sizes, int n_in,
                              void* d_out, int out_size, void* d_ws, size_t ws_size,
                              hipStream_t stream) {
  auto F = [&](int i) { return (const float*)d_in[i]; };
  const float* node_sca = F(0);
  const float* node_vec = F(1);
  const float* pos      = F(2);
  const int*   eidx     = (const int*)d_in[3];
  const float* eeW      = F(4);
  const float* ngv_wv1 = F(5),  *ngv_wv2 = F(6),  *ngv_ws = F(7),  *ngv_wg = F(8),  *ngv_bg = F(9);
  const float* egv_wv1 = F(10), *egv_wv2 = F(11), *egv_ws = F(12), *egv_wg = F(13), *egv_bg = F(14);
  const float* ogv_wv1 = F(15), *ogv_wv2 = F(16), *ogv_ws = F(17), *ogv_wg = F(18), *ogv_bg = F(19);
  const float* cgv_wv1 = F(20), *cgv_wv2 = F(21), *cgv_ws = F(22), *cgv_wg = F(23), *cgv_bg = F(24);
  const float* tgv_wv1 = F(25), *tgv_wv2 = F(26), *tgv_ws = F(27), *tgv_wg = F(28), *tgv_bg = F(29);
  const float* egv_dir = F(30);
  const float* msca_w = F(31), *msca_b = F(32);
  const float* e2n_w  = F(33), *e2n_b  = F(34);
  const float* n2e_w  = F(35), *n2e_b  = F(36);
  const float* evn_w  = F(37), *evn_b  = F(38);
  const float* act_dir = F(39);
  const float* ln_sg = F(40), *ln_sb = F(41), *ln_vg = F(42), *ln_vb = F(43);

  const int N = in_sizes[0] / SCA;
  const int E = in_sizes[3] / 2;
  const int* row = eidx;
  const int* col = eidx + E;

  float* p = (float*)d_ws;
  auto alloc = [&](size_t n) { float* r = p; p += n; return r; };
  float* sca_cur = alloc((size_t)N * 256);
  float* vecT    = alloc((size_t)N * 192);
  float* nsca    = alloc((size_t)N * 256);
  float* nvecT   = alloc((size_t)N * 192);
  float* csca    = alloc((size_t)N * 256);
  float* cvecT   = alloc((size_t)N * 192);
  float* osca    = alloc((size_t)N * 256);
  float* ovecT   = alloc((size_t)N * 192);
  float* dsca    = alloc((size_t)N * 256);
  float* dvecT   = alloc((size_t)N * 192);
  float* a_sca   = alloc((size_t)N * 256);
  float* a_vecT  = alloc((size_t)N * 192);
  float* vi_n    = alloc((size_t)N * 192);
  float* cat_n   = alloc((size_t)N * 320);
  float* gate_n  = alloc((size_t)N * 64);
  float* dv_n    = alloc((size_t)N * 192);
  float* prep    = alloc(192);
  float* e_sca   = alloc((size_t)E * 64);
  float* unit    = alloc((size_t)E * 3);
  float* Cc      = alloc((size_t)E);
  float* un      = alloc((size_t)E);
  float* s2v     = alloc((size_t)E);
  float* esca    = alloc((size_t)E * 64);
  float* gate_e  = alloc((size_t)E * 64);
  float* gbuf    = alloc((size_t)E * 64);
  float* ddbuf   = alloc((size_t)E * 64);
  float* fbuf    = alloc((size_t)E * 64);
  float* t2      = alloc((size_t)E * 64);
  float* t3      = alloc((size_t)E * 64);
  float* t4      = alloc((size_t)E * 64);
  float* y_sca   = alloc((size_t)E * 256);
  float* y_vecT  = alloc((size_t)E * 192);
  float* vi_e    = alloc((size_t)E * 192);
  float* cat_e   = alloc((size_t)E * 320);
  float* gate_oe = alloc((size_t)E * 64);
  float* msca0   = alloc((size_t)E * 256);
  float* mvecT   = alloc((size_t)E * 192);

  // ---- init state + edge geometry (layer-invariant) ----
  k_copy<<<cdiv(N * 256, 256), 256, 0, stream>>>(node_sca, sca_cur, N * 256);
  k_vec_in<<<cdiv(N * 192, 256), 256, 0, stream>>>(node_vec, vecT, N);
  k_edge_geom<<<cdiv(E, 256), 256, 0, stream>>>(eidx, pos, E, unit, Cc, un, s2v, e_sca);

  for (int l = 0; l < NLAYER; ++l) {
    const float* Lngv_wv1 = ngv_wv1 + (size_t)l * 64 * 64;
    const float* Lngv_wv2 = ngv_wv2 + (size_t)l * 64 * 64;
    const float* Lngv_ws  = ngv_ws  + (size_t)l * 256 * 320;
    const float* Lngv_wg  = ngv_wg  + (size_t)l * 64 * 256;
    const float* Lngv_bg  = ngv_bg  + (size_t)l * 64;
    const float* Legv_ws  = egv_ws  + (size_t)l * 64 * 128;
    const float* Legv_wg  = egv_wg  + (size_t)l * 64 * 64;
    const float* Legv_bg  = egv_bg  + (size_t)l * 64;
    const float* Logv_wv1 = ogv_wv1 + (size_t)l * 64 * 64;
    const float* Logv_wv2 = ogv_wv2 + (size_t)l * 64 * 64;
    const float* Logv_ws  = ogv_ws  + (size_t)l * 256 * 320;
    const float* Logv_wg  = ogv_wg  + (size_t)l * 64 * 256;
    const float* Logv_bg  = ogv_bg  + (size_t)l * 64;
    const float* Lcgv_wv1 = cgv_wv1 + (size_t)l * 64 * 64;
    const float* Lcgv_wv2 = cgv_wv2 + (size_t)l * 64 * 64;
    const float* Lcgv_ws  = cgv_ws  + (size_t)l * 256 * 320;
    const float* Lcgv_wg  = cgv_wg  + (size_t)l * 64 * 256;
    const float* Lcgv_bg  = cgv_bg  + (size_t)l * 64;
    const float* Ltgv_wv1 = tgv_wv1 + (size_t)l * 64 * 64;
    const float* Ltgv_wv2 = tgv_wv2 + (size_t)l * 64 * 64;
    const float* Ltgv_ws  = tgv_ws  + (size_t)l * 256 * 320;
    const float* Ltgv_wg  = tgv_wg  + (size_t)l * 64 * 256;
    const float* Ltgv_bg  = tgv_bg  + (size_t)l * 64;

    // fold rank-1 edge-expansion structure for this layer
    k_layer_prep<<<1, 64, 0, stream>>>(egv_wv1 + (size_t)l * 64 * 64,
                                       egv_wv2 + (size_t)l * 64 * 64,
                                       eeW + (size_t)l * 64, Legv_ws, prep);

    // node message features: ngv gvlinear
    gvlinear(stream, sca_cur, vecT, N, Lngv_wv1, Lngv_wv2, Lngv_ws, Lngv_wg, Lngv_bg,
             vi_n, cat_n, gate_n, nsca, nvecT);

    // edge scalar chain: esca_raw = A*||u|| + e_sca @ ws2.T
    gemm(stream, e_sca, 64, nullptr, Legv_ws + 64, 128, 64, nullptr,
         prep + 128, un, 0, nullptr, nullptr, 1, 0, esca, 64, E, 64);
    gemm(stream, esca, 64, nullptr, Legv_wg, 64, 64, Legv_bg, nullptr, nullptr, 2,
         nullptr, nullptr, 1, 0, gate_e, 64, E, 64);
    k_gmul<<<cdiv(E * 64, 256), 256, 0, stream>>>(gate_e, prep, gbuf, E * 64);
    gemm(stream, gbuf, 64, nullptr, egv_dir + (size_t)l * 64 * 64, 64, 64, nullptr,
         nullptr, nullptr, 0, nullptr, nullptr, 1, 0, ddbuf, 64, E, 64);
    k_f_lrelu<<<cdiv(E * 64, 256), 256, 0, stream>>>(esca, gbuf, ddbuf, s2v, fbuf, E);

    // messages
    gemm(stream, esca, 64, nullptr, msca_w + (size_t)l * 256 * 64, 64, 64,
         msca_b + (size_t)l * 256, nullptr, nullptr, 0,
         nsca, col, 1, 256, y_sca, 256, E, 256);
    gemm(stream, esca, 64, nullptr, e2n_w + (size_t)l * 64 * 64, 64, 64,
         e2n_b + (size_t)l * 64, nullptr, nullptr, 0,
         nullptr, nullptr, 1, 0, t2, 64, E, 64);
    gemm(stream, nsca, 256, col, n2e_w + (size_t)l * 64 * 256, 256, 256,
         n2e_b + (size_t)l * 64, nullptr, nullptr, 0,
         nullptr, nullptr, 1, 0, t3, 64, E, 64);
    gemm(stream, fbuf, 64, nullptr, evn_w + (size_t)l * 64 * 64, 64, 64, nullptr,
         nullptr, nullptr, 0, nullptr, nullptr, 1, 0, t4, 64, E, 64);
    k_yvec<<<cdiv(E * 64, 256), 256, 0, stream>>>(t2, t3, t4, nvecT, unit,
                                                  evn_b + (size_t)l * 64, col, y_vecT, E);

    // output gvlinear on edges
    gvlinear(stream, y_sca, y_vecT, E, Logv_wv1, Logv_wv2, Logv_ws, Logv_wg, Logv_bg,
             vi_e, cat_e, gate_oe, msca0, mvecT);

    // aggregate (scatter-sum with cosine cutoff)
    k_zero<<<cdiv(N * 256, 256), 256, 0, stream>>>(a_sca, N * 256);
    k_zero<<<cdiv(N * 192, 256), 256, 0, stream>>>(a_vecT, N * 192);
    k_scatter_sca<<<cdiv(E * 256, 256), 256, 0, stream>>>(msca0, Cc, row, a_sca, E);
    k_scatter_vec<<<cdiv(E * 192, 256), 256, 0, stream>>>(mvecT, Cc, row, a_vecT, E);

    // cgv + combine
    gvlinear(stream, sca_cur, vecT, N, Lcgv_wv1, Lcgv_wv2, Lcgv_ws, Lcgv_wg, Lcgv_bg,
             vi_n, cat_n, gate_n, csca, cvecT);
    k_combine_sca<<<N, 256, 0, stream>>>(csca, a_sca, ln_sg + (size_t)l * 256,
                                         ln_sb + (size_t)l * 256, osca);
    k_combine_vec<<<N, 192, 0, stream>>>(cvecT, a_vecT, ln_vg + (size_t)l * 192,
                                         ln_vb + (size_t)l * 192, ovecT);
    gemm(stream, ovecT, 64, nullptr, act_dir + (size_t)l * 64 * 64, 64, 64, nullptr,
         nullptr, nullptr, 0, nullptr, nullptr, 1, 0, dv_n, 64, N * 3, 64);
    k_vnlrelu<<<cdiv(N * 64, 256), 256, 0, stream>>>(ovecT, dv_n, N);

    // tgv + residual
    gvlinear(stream, osca, ovecT, N, Ltgv_wv1, Ltgv_wv2, Ltgv_ws, Ltgv_wg, Ltgv_bg,
             vi_n, cat_n, gate_n, dsca, dvecT);
    k_add<<<cdiv(N * 256, 256), 256, 0, stream>>>(sca_cur, dsca, N * 256);
    k_add<<<cdiv(N * 192, 256), 256, 0, stream>>>(vecT, dvecT, N * 192);
  }

  k_out_pack<<<cdiv(N * (256 + 192), 256), 256, 0, stream>>>(sca_cur, vecT, (float*)d_out, N);
}
